// SimplifiedNGPNeRF_44985487458774
// MI455X (gfx1250) — matrix operand order repaired and checked by reading, not compiled, not run on hardware
//
#include <hip/hip_runtime.h>
#include <hip/hip_bf16.h>

#define NPTS 262144
#define TSIZE 524288u
#define TMASK 524287u

typedef _Float16 v16h __attribute__((ext_vector_type(16)));
typedef float    v8f  __attribute__((ext_vector_type(8)));

union V16HU { v16h v; uint4 q[2]; };

// Per-level grid resolutions: floor(16 * (128^(1/15))^l)
__constant__ float NLVL[16] = {16.f, 22.f, 30.f, 42.f, 58.f, 80.f, 111.f, 153.f,
                               212.f, 294.f, 406.f, 561.f, 776.f, 1072.f, 1481.f, 2048.f};

static __device__ __forceinline__ v8f wmma_f16(v16h a, v16h b, v8f c) {
    // D = A(16x32 f16) * B(32x16 f16) + C(16x16 f32)
    return __builtin_amdgcn_wmma_f32_16x16x32_f16(false, a, false, b, (short)0, c, false, false);
}

// Load one 16x32 f16 A-operand K-chunk from a row-major f16 row pointer.
// Documented 16-bit A layout: lanes 0-15 hold K 0..7 & 16..23, lanes 16-31 hold K 8..15 & 24..31.
static __device__ __forceinline__ v16h ld_a(const _Float16* rowp, int hi, int chunk) {
    V16HU t;
    const char* p = (const char*)rowp + chunk * 64 + hi * 16;
    t.q[0] = *(const uint4*)(p);        // K = chunk*32 + hi*8 .. +7
    t.q[1] = *(const uint4*)(p + 32);   // K = chunk*32 + 16 + hi*8 .. +7
    return t.v;
}

// Load one pre-packed 32x16 f16 B tile (lane-major, 16 halves per lane).
static __device__ __forceinline__ v16h ld_b(const _Float16* __restrict__ wB, int tile, int lane) {
    V16HU t;
    const uint4* p = (const uint4*)(wB + tile * 512 + lane * 16);
    t.q[0] = p[0];
    t.q[1] = p[1];
    return t.v;
}

// ---------------------------------------------------------------------------
// Repack f32 weights into WMMA B-operand f16 tiles.
// B layout (K=32 tile): lane l holds column n=l&15, K = (l>>4)*8 + h (h<8), 16 + (l>>4)*8 + (h-8) (h>=8).
// Tiles: 0..3  dW1 (32x64, nt)       4..5  dW2 (64x16, kc)
//        6..13 cW1 (43->64 x64)      14..21 cW2 (64x64)
// ---------------------------------------------------------------------------
__global__ __launch_bounds__(256) void prep_weights(
    const float* __restrict__ dW1, const float* __restrict__ dW2,
    const float* __restrict__ cW1, const float* __restrict__ cW2,
    _Float16* __restrict__ wB)
{
    int t = blockIdx.x * 256 + threadIdx.x;
    if (t >= 22 * 512) return;
    int tile = t >> 9;
    int e = t & 511;
    int lane = e >> 4;
    int h = e & 15;
    int kl = (h < 8) ? ((lane >> 4) * 8 + h) : (16 + (lane >> 4) * 8 + (h - 8));   // K within 32-chunk
    int nl = lane & 15;                // column within 16-tile
    float v = 0.f;
    if (tile < 4) {                      // dW1: K=32, N=64
        v = dW1[kl * 64 + (tile * 16 + nl)];
    } else if (tile < 6) {               // dW2: K=64, N=16
        int k = (tile - 4) * 32 + kl;
        v = dW2[k * 16 + nl];
    } else if (tile < 14) {              // cW1: K=43 (pad 64), N=64
        int idx = tile - 6;
        int k = (idx >> 2) * 32 + kl;
        int n = (idx & 3) * 16 + nl;
        v = (k < 43) ? cW1[k * 64 + n] : 0.f;
    } else {                             // cW2: K=64, N=64
        int idx = tile - 14;
        int k = (idx >> 2) * 32 + kl;
        int n = (idx & 3) * 16 + nl;
        v = cW2[k * 64 + n];
    }
    wB[t] = (_Float16)v;
}

// ---------------------------------------------------------------------------
// Hash-grid encode + direction PE + AABB mask. One thread per point.
// Gathers are random 8B reads into a 64MB table -> L2 resident on MI455X.
// ---------------------------------------------------------------------------
__global__ __launch_bounds__(256) void ngp_encode(
    const float* __restrict__ x, const float* __restrict__ d,
    const float* __restrict__ tables,
    _Float16* __restrict__ featsA, _Float16* __restrict__ denc,
    unsigned char* __restrict__ maskp)
{
    int i = blockIdx.x * 256 + threadIdx.x;
    if (i >= NPTS) return;

    float px = x[i * 3 + 0] * (1.f / 3.f);
    float py = x[i * 3 + 1] * (1.f / 3.f);
    float pz = x[i * 3 + 2] * (1.f / 3.f);
    bool m = (fabsf(px) < 0.5f) && (fabsf(py) < 0.5f) && (fabsf(pz) < 0.5f);
    maskp[i] = m ? 1 : 0;
    px += 0.5f; py += 0.5f; pz += 0.5f;

    const float2* tab = (const float2*)tables;
    _Float16* fp = featsA + (size_t)i * 32;

    for (int l = 0; l < 16; ++l) {
        float nlev = NLVL[l];
        float sx = px * nlev, sy = py * nlev, sz = pz * nlev;
        float fx = floorf(sx), fy = floorf(sy), fz = floorf(sz);
        float lx = sx - fx, ly = sy - fy, lz = sz - fz;
        int ix = (int)fx, iy = (int)fy, iz = (int)fz;
        float f0 = 0.f, f1 = 0.f;
        const float2* tl = tab + (size_t)l * TSIZE;
#pragma unroll
        for (int cr = 0; cr < 8; ++cr) {
            int ox = cr & 1, oy = (cr >> 1) & 1, oz = (cr >> 2) & 1;
            unsigned vx = (unsigned)(ix + ox);
            unsigned vy = (unsigned)(iy + oy);
            unsigned vz = (unsigned)(iz + oz);
            unsigned hh = (vx ^ (vy * 2654435761u) ^ (vz * 805459861u)) & TMASK;
            float w = (ox ? lx : 1.f - lx) * (oy ? ly : 1.f - ly) * (oz ? lz : 1.f - lz);
            float2 tv = tl[hh];
            f0 += w * tv.x;
            f1 += w * tv.y;
        }
        fp[2 * l + 0] = (_Float16)f0;
        fp[2 * l + 1] = (_Float16)f1;
    }

    // direction positional encoding -> 27 values, zero-padded to 48
    float dx = d[i * 3 + 0], dy = d[i * 3 + 1], dz = d[i * 3 + 2];
    _Float16* dp = denc + (size_t)i * 48;
    dp[0] = (_Float16)dx; dp[1] = (_Float16)dy; dp[2] = (_Float16)dz;
    int o = 3;
#pragma unroll
    for (int j = 0; j < 4; ++j) {
        float s = (float)(1 << j);
        dp[o + 0] = (_Float16)sinf(s * dx);
        dp[o + 1] = (_Float16)sinf(s * dy);
        dp[o + 2] = (_Float16)sinf(s * dz);
        dp[o + 3] = (_Float16)cosf(s * dx);
        dp[o + 4] = (_Float16)cosf(s * dy);
        dp[o + 5] = (_Float16)cosf(s * dz);
        o += 6;
    }
#pragma unroll
    for (int j = 27; j < 48; ++j) dp[j] = (_Float16)0.f;
}

// ---------------------------------------------------------------------------
// Fused MLP: 256 threads = 8 waves; each wave owns a 16-point row tile.
// All four hidden GEMM layers run on v_wmma_f32_16x16x32_f16; activations
// transit through per-wave LDS buffers between layers.
// ---------------------------------------------------------------------------
__global__ __launch_bounds__(256) void ngp_mlp(
    const _Float16* __restrict__ featsA, const _Float16* __restrict__ denc,
    const unsigned char* __restrict__ maskp, const _Float16* __restrict__ wB,
    const float* __restrict__ db1, const float* __restrict__ db2,
    const float* __restrict__ cb1, const float* __restrict__ cb2,
    const float* __restrict__ cW3, const float* __restrict__ cb3,
    float* __restrict__ outRgb, float* __restrict__ outSigma)
{
    __shared__ __align__(16) _Float16 sb[8][2][16][64];   // 32 KB

    const int tid = threadIdx.x;
    const int wave = tid >> 5;
    const int lane = tid & 31;
    const int ln = lane & 15;     // column index within 16-tile (D layout)
    const int hi = lane >> 4;     // row-half selector
    const int rowBase = blockIdx.x * 128 + wave * 16;

    // ---- density layer 1: [16x32] @ [32x64] -> relu -> LDS buf0 -------------
    v16h aF = ld_a(featsA + (size_t)(rowBase + ln) * 32, hi, 0);
#pragma unroll
    for (int nt = 0; nt < 4; ++nt) {
        v8f c = {};
        c = wmma_f16(aF, ld_b(wB, nt, lane), c);
        float bias = db1[nt * 16 + ln];
#pragma unroll
        for (int r = 0; r < 8; ++r) {
            float v = c[r] + bias;
            v = v > 0.f ? v : 0.f;
            sb[wave][0][hi * 8 + r][nt * 16 + ln] = (_Float16)v;
        }
    }
    __syncthreads();

    // ---- density layer 2: [16x64] @ [64x16] -> h ----------------------------
    v8f hc = {};
    hc = wmma_f16(ld_a(&sb[wave][0][ln][0], hi, 0), ld_b(wB, 4, lane), hc);
    hc = wmma_f16(ld_a(&sb[wave][0][ln][0], hi, 1), ld_b(wB, 5, lane), hc);
    float b2 = db2[ln];
    float hv[8];
#pragma unroll
    for (int r = 0; r < 8; ++r) hv[r] = hc[r] + b2;

    // sigma = exp(where(mask, h[:,0], -1e5)); lanes with ln==0 hold column 0
    if (ln == 0) {
#pragma unroll
        for (int r = 0; r < 8; ++r) {
            int g = rowBase + hi * 8 + r;
            outSigma[g] = maskp[g] ? expf(hv[r]) : 0.f;
        }
    }

    // ---- assemble color input in buf1: [h(16) | d_enc(27) | pad] -----------
#pragma unroll
    for (int r = 0; r < 8; ++r)
        sb[wave][1][hi * 8 + r][ln] = (_Float16)hv[r];
#pragma unroll
    for (int t = 0; t < 3; ++t) {   // copy 16 rows x 48 halves (96 x 16B chunks)
        int id = t * 32 + lane;
        int rr = id / 6, oo = id % 6;
        *(uint4*)&sb[wave][1][rr][16 + oo * 8] =
            *(const uint4*)(denc + (size_t)(rowBase + rr) * 48 + oo * 8);
    }
    __syncthreads();

    // ---- color layer 1: [16x64] @ [64x64] -> relu -> buf0 -------------------
#pragma unroll
    for (int nt = 0; nt < 4; ++nt) {
        v8f c = {};
#pragma unroll
        for (int kc = 0; kc < 2; ++kc)
            c = wmma_f16(ld_a(&sb[wave][1][ln][0], hi, kc), ld_b(wB, 6 + kc * 4 + nt, lane), c);
        float bias = cb1[nt * 16 + ln];
#pragma unroll
        for (int r = 0; r < 8; ++r) {
            float v = c[r] + bias;
            v = v > 0.f ? v : 0.f;
            sb[wave][0][hi * 8 + r][nt * 16 + ln] = (_Float16)v;
        }
    }
    __syncthreads();

    // ---- color layer 2: [16x64] @ [64x64] -> relu -> buf1 -------------------
#pragma unroll
    for (int nt = 0; nt < 4; ++nt) {
        v8f c = {};
#pragma unroll
        for (int kc = 0; kc < 2; ++kc)
            c = wmma_f16(ld_a(&sb[wave][0][ln][0], hi, kc), ld_b(wB, 14 + kc * 4 + nt, lane), c);
        float bias = cb2[nt * 16 + ln];
#pragma unroll
        for (int r = 0; r < 8; ++r) {
            float v = c[r] + bias;
            v = v > 0.f ? v : 0.f;
            sb[wave][1][hi * 8 + r][nt * 16 + ln] = (_Float16)v;
        }
    }
    __syncthreads();

    // ---- output layer 64 -> 3, sigmoid, mask --------------------------------
    if (lane < 16) {
        int g = rowBase + lane;
        float a0 = cb3[0], a1 = cb3[1], a2 = cb3[2];
#pragma unroll
        for (int k = 0; k < 64; ++k) {
            float v = (float)sb[wave][1][lane][k];
            a0 += v * cW3[k * 3 + 0];
            a1 += v * cW3[k * 3 + 1];
            a2 += v * cW3[k * 3 + 2];
        }
        bool m = maskp[g] != 0;
        outRgb[g * 3 + 0] = m ? 1.f / (1.f + expf(-a0)) : 0.f;
        outRgb[g * 3 + 1] = m ? 1.f / (1.f + expf(-a1)) : 0.f;
        outRgb[g * 3 + 2] = m ? 1.f / (1.f + expf(-a2)) : 0.f;
    }
}

extern "C" void kernel_launch(void* const* d_in, const int* in_sizes, int n_in,
                              void* d_out, int out_size, void* d_ws, size_t ws_size,
                              hipStream_t stream) {
    const int N = NPTS;
    const float* x      = (const float*)d_in[0];
    const float* d      = (const float*)d_in[1];
    const float* tables = (const float*)d_in[2];
    const float* dW1 = (const float*)d_in[3];  const float* db1 = (const float*)d_in[4];
    const float* dW2 = (const float*)d_in[5];  const float* db2 = (const float*)d_in[6];
    const float* cW1 = (const float*)d_in[7];  const float* cb1 = (const float*)d_in[8];
    const float* cW2 = (const float*)d_in[9];  const float* cb2 = (const float*)d_in[10];
    const float* cW3 = (const float*)d_in[11]; const float* cb3 = (const float*)d_in[12];

    char* ws = (char*)d_ws;
    _Float16* wB     = (_Float16*)ws;                                        // 22.5 KB
    _Float16* featsA = (_Float16*)(ws + 32768);                              // N*64 B
    _Float16* denc   = (_Float16*)(ws + 32768 + (size_t)N * 64);             // N*96 B
    unsigned char* maskp = (unsigned char*)(ws + 32768 + (size_t)N * 160);   // N B

    float* outRgb = (float*)d_out;
    float* outSigma = outRgb + (size_t)N * 3;

    prep_weights<<<44, 256, 0, stream>>>(dW1, dW2, cW1, cW2, wB);
    ngp_encode<<<N / 256, 256, 0, stream>>>(x, d, tables, featsA, denc, maskp);
    ngp_mlp<<<N / 128, 256, 0, stream>>>(featsA, denc, maskp, wB,
                                         db1, db2, cb1, cb2, cW3, cb3,
                                         outRgb, outSigma);
}
